// WindowAttention_15564961481315
// MI455X (gfx1250) — matrix operand order repaired and checked by reading, compiled, not run on hardware
//
#include <hip/hip_runtime.h>

// ---------------------------------------------------------------------------
// Shifted-window attention for MI455X (gfx1250), bf16 WMMA path.
//   prep:     weight transpose->bf16 [N][K], 64x64 relative-position bias
//   win_attn: fused QKV + biased/masked softmax attention per 8x8 window
//   out_proj: final 192x192 projection + bias + inverse roll
// Workspace: wsW 221184B | wsWo 73728B | bias 16384B | att bf16 50331648B
// ---------------------------------------------------------------------------

typedef __bf16 bf16;
typedef __attribute__((ext_vector_type(16))) bf16 v16bf;
typedef __attribute__((ext_vector_type(8)))  bf16 v8bf;
typedef __attribute__((ext_vector_type(8)))  float v8f;

__device__ __forceinline__ v8f wmma_bf16(v16bf a, v16bf b, v8f c) {
  return __builtin_amdgcn_wmma_f32_16x16x32_bf16(false, a, false, b, (short)0, c,
                                                 false, false);
}

// A-fragment (16x32 bf16): caller passes row base + (lane>=16 ? 8 : 0).
// Lane holds K = base..base+7 and base+16..base+23.
__device__ __forceinline__ v16bf load_fragA(const bf16* p) {
  v8bf lo = *(const v8bf*)p;
  v8bf hi = *(const v8bf*)(p + 16);
  v16bf r;
#pragma unroll
  for (int i = 0; i < 8; ++i) { r[i] = lo[i]; r[i + 8] = hi[i]; }
  return r;
}

// B-fragment (32x16 bf16, pre-transposed [N][K]): caller passes column base
// + (lane>=16 ? 8 : 0). Lane holds K = base..base+7 and base+16..base+23.
__device__ __forceinline__ v16bf load_fragB(const bf16* p) {
  v8bf lo = *(const v8bf*)p;
  v8bf hi = *(const v8bf*)(p + 16);
  v16bf r;
#pragma unroll
  for (int i = 0; i < 8; ++i) { r[i] = lo[i]; r[i + 8] = hi[i]; }
  return r;
}

// ---------------------------------------------------------------------------
__global__ void prep_kernel(const float* __restrict__ w_qkv,
                            const float* __restrict__ w_out,
                            const float* __restrict__ pos,
                            bf16* __restrict__ wsW, bf16* __restrict__ wsWo,
                            float* __restrict__ wsBias) {
  int idx = blockIdx.x * blockDim.x + threadIdx.x;
  int stride = gridDim.x * blockDim.x;
  for (int i = idx; i < 576 * 192; i += stride) {
    int n = i / 192, k = i % 192;
    wsW[i] = (bf16)w_qkv[k * 576 + n];
  }
  for (int i = idx; i < 192 * 192; i += stride) {
    int n = i / 192, k = i % 192;
    wsWo[i] = (bf16)w_out[k * 192 + n];
  }
  for (int i = idx; i < 4096; i += stride) {
    int ii = i >> 6, jj = i & 63;
    int iy = ii >> 3, ix = ii & 7, jy = jj >> 3, jx = jj & 7;
    wsBias[i] = pos[(jy - iy + 7) * 15 + (jx - ix + 7)];
  }
}

// ---------------------------------------------------------------------------
#define XST 200  // x-tile row stride (bf16) : 400B, 16B aligned, low-conflict
#define QST 40   // Q/K row stride          : 80B
#define VST 72   // V^T row stride          : 144B
#define PST 72   // P row stride            : 144B

__global__ void __launch_bounds__(128)
win_attn_kernel(const float* __restrict__ x, const bf16* __restrict__ wsW,
                const float* __restrict__ wsBias, bf16* __restrict__ att) {
  __shared__ bf16 xtile[64 * XST];  // 25600 B
  __shared__ bf16 Qs[64 * QST];     //  5120 B
  __shared__ bf16 Ks[64 * QST];     //  5120 B
  __shared__ bf16 Vt[32 * VST];     //  4608 B
  __shared__ bf16 Ps[64 * PST];     //  9216 B

  const int tid = threadIdx.x;
  const int lane = tid & 31;
  const int wv = tid >> 5;
  const int l15 = lane & 15;
  const int half = lane >> 4;
  const int abase = half * 8;
  const int bbase = half * 8;

  const int blk = blockIdx.x;         // 2048 blocks: 8 batches x 256 windows
  const int b = blk >> 8;
  const int w = blk & 255;
  const int wy = w >> 4, wx = w & 15;
  const bool ulact = (wy == 15);
  const bool lract = (wx == 15);

  // Stage rolled 64x192 window (f32 -> bf16) in LDS. Roll folded into index.
  for (int t = tid; t < 64 * 48; t += 128) {
    int tok = t / 48, c4 = t % 48;
    int gy = (wy * 8 + (tok >> 3) + 4) & 127;
    int gx = (wx * 8 + (tok & 7) + 4) & 127;
    const float4 v =
        *(const float4*)(x + (((long)b * 128 + gy) * 128 + gx) * 192 + c4 * 4);
    bf16* dst = &xtile[tok * XST + c4 * 4];
    dst[0] = (bf16)v.x; dst[1] = (bf16)v.y; dst[2] = (bf16)v.z; dst[3] = (bf16)v.w;
  }
  __syncthreads();

  const float scale = 0.17677669529663687f;  // 32^-0.5 folded into Q
  const int mrow = wv * 16 + l15;            // A-frag row for this lane

  for (int h = 0; h < 6; ++h) {
    // ---- QKV for this head: wave computes rows [16*wv,16*wv+16) x 96 ----
    v8f qa[2] = {}, ka[2] = {}, va[2] = {};
#pragma unroll
    for (int kk = 0; kk < 6; ++kk) {
      v16bf a = load_fragA(&xtile[mrow * XST + kk * 32 + abase]);
#pragma unroll
      for (int dt = 0; dt < 2; ++dt) {
        int n = h * 32 + dt * 16 + l15;
        v16bf bq = load_fragB(wsW + (long)(0 + n) * 192 + kk * 32 + bbase);
        qa[dt] = wmma_bf16(a, bq, qa[dt]);
        v16bf bk = load_fragB(wsW + (long)(192 + n) * 192 + kk * 32 + bbase);
        ka[dt] = wmma_bf16(a, bk, ka[dt]);
        v16bf bv = load_fragB(wsW + (long)(384 + n) * 192 + kk * 32 + bbase);
        va[dt] = wmma_bf16(a, bv, va[dt]);
      }
    }
#pragma unroll
    for (int dt = 0; dt < 2; ++dt) {
#pragma unroll
      for (int g = 0; g < 8; ++g) {
        int row = wv * 16 + g + half * 8;  // C layout: VGPR g <-> row
        int col = dt * 16 + l15;
        Qs[row * QST + col] = (bf16)(qa[dt][g] * scale);
        Ks[row * QST + col] = (bf16)ka[dt][g];
        Vt[col * VST + row] = (bf16)va[dt][g];  // store V transposed
      }
    }
    __syncthreads();

    // ---- S = Qs K^T + bias (+ shift masks), softmax in registers ----
    v16bf aq = load_fragA(&Qs[mrow * QST + abase]);
    v8f s[4];
    v8f zro = {};
#pragma unroll
    for (int nt = 0; nt < 4; ++nt) {
      v16bf bk = load_fragB(&Ks[(nt * 16 + l15) * QST + bbase]);
      s[nt] = wmma_bf16(aq, bk, zro);
    }
#pragma unroll
    for (int nt = 0; nt < 4; ++nt) {
#pragma unroll
      for (int g = 0; g < 8; ++g) {
        int i = wv * 16 + g + half * 8;
        int j = nt * 16 + l15;
        float v = s[nt][g] + wsBias[i * 64 + j];
        if (ulact && ((i >= 32) != (j >= 32))) v = -3.0e38f;
        if (lract && (((i & 7) >= 4) != ((j & 7) >= 4))) v = -3.0e38f;
        s[nt][g] = v;
      }
    }
    float rinv[8];
#pragma unroll
    for (int g = 0; g < 8; ++g) {  // rows of a fixed g live in one 16-lane half
      float m = fmaxf(fmaxf(s[0][g], s[1][g]), fmaxf(s[2][g], s[3][g]));
      m = fmaxf(m, __shfl_xor(m, 1));
      m = fmaxf(m, __shfl_xor(m, 2));
      m = fmaxf(m, __shfl_xor(m, 4));
      m = fmaxf(m, __shfl_xor(m, 8));
      float t = 0.f;
#pragma unroll
      for (int nt = 0; nt < 4; ++nt) {
        float e = __expf(s[nt][g] - m);
        s[nt][g] = e;
        t += e;
      }
      t += __shfl_xor(t, 1);
      t += __shfl_xor(t, 2);
      t += __shfl_xor(t, 4);
      t += __shfl_xor(t, 8);
      rinv[g] = 1.0f / t;
    }
#pragma unroll
    for (int nt = 0; nt < 4; ++nt)
#pragma unroll
      for (int g = 0; g < 8; ++g) {
        int i = wv * 16 + g + half * 8;
        Ps[i * PST + nt * 16 + l15] = (bf16)s[nt][g];  // same-wave rows only
      }

    // ---- out = (P V) * rinv ; rows match this wave, no barrier needed ----
    v8f o[2] = {};
#pragma unroll
    for (int kk = 0; kk < 2; ++kk) {
      v16bf ap = load_fragA(&Ps[mrow * PST + kk * 32 + abase]);
#pragma unroll
      for (int dt = 0; dt < 2; ++dt) {
        v16bf bv = load_fragB(&Vt[(dt * 16 + l15) * VST + kk * 32 + bbase]);
        o[dt] = wmma_bf16(ap, bv, o[dt]);
      }
    }
#pragma unroll
    for (int dt = 0; dt < 2; ++dt)
#pragma unroll
      for (int g = 0; g < 8; ++g) {
        int i = wv * 16 + g + half * 8;
        int gy = wy * 8 + (i >> 3);
        int gx = wx * 8 + (i & 7);
        long tok = (long)b * 16384 + gy * 128 + gx;  // rolled coordinates
        att[tok * 192 + h * 32 + dt * 16 + l15] = (bf16)(o[dt][g] * rinv[g]);
      }
    __syncthreads();  // before next head reuses Qs/Ks/Vt
  }
}

// ---------------------------------------------------------------------------
__global__ void __launch_bounds__(128)
out_proj_kernel(const bf16* __restrict__ att, const bf16* __restrict__ wsWo,
                const float* __restrict__ b_out, float* __restrict__ out) {
  const int tid = threadIdx.x;
  const int lane = tid & 31;
  const int wv = tid >> 5;
  const int l15 = lane & 15;
  const int half = lane >> 4;
  const int abase = half * 8;
  const int bbase = half * 8;

  const long mb = (long)blockIdx.x * 64 + (wv >> 1) * 32;
  const int nb = blockIdx.y * 64 + (wv & 1) * 32;

  v8f c[2][2] = {};
#pragma unroll
  for (int kk = 0; kk < 6; ++kk) {
    v16bf a0 = load_fragA(att + (mb + l15) * 192 + kk * 32 + abase);
    v16bf a1 = load_fragA(att + (mb + 16 + l15) * 192 + kk * 32 + abase);
#pragma unroll
    for (int ni = 0; ni < 2; ++ni) {
      v16bf bb = load_fragB(wsWo + (long)(nb + ni * 16 + l15) * 192 + kk * 32 + bbase);
      c[0][ni] = wmma_bf16(a0, bb, c[0][ni]);
      c[1][ni] = wmma_bf16(a1, bb, c[1][ni]);
    }
  }
#pragma unroll
  for (int mi = 0; mi < 2; ++mi)
#pragma unroll
    for (int ni = 0; ni < 2; ++ni) {
      int n = nb + ni * 16 + l15;
      float bias = b_out[n];
#pragma unroll
      for (int g = 0; g < 8; ++g) {
        long tok = mb + mi * 16 + g + half * 8;
        int bsel = (int)(tok >> 14);
        int y = (int)((tok >> 7) & 127);
        int xx = (int)(tok & 127);
        long dst = ((long)bsel << 14) + (long)(((y + 4) & 127) << 7) +
                   ((xx + 4) & 127);                       // inverse roll
        out[dst * 192 + n] = c[mi][ni][g] + bias;
      }
    }
}

// ---------------------------------------------------------------------------
extern "C" void kernel_launch(void* const* d_in, const int* in_sizes, int n_in,
                              void* d_out, int out_size, void* d_ws,
                              size_t ws_size, hipStream_t stream) {
  const float* x = (const float*)d_in[0];
  const float* w_qkv = (const float*)d_in[1];
  const float* pos = (const float*)d_in[2];
  const float* w_out = (const float*)d_in[3];
  const float* b_out = (const float*)d_in[4];
  float* out = (float*)d_out;

  char* ws = (char*)d_ws;
  bf16* wsW = (bf16*)ws;                      // 576*192 bf16 = 221184 B
  bf16* wsWo = (bf16*)(ws + 221184);          // 192*192 bf16 =  73728 B
  float* wsBias = (float*)(ws + 294912);      // 64*64 f32    =  16384 B
  bf16* wsAtt = (bf16*)(ws + 311296);         // 8*128*128*192 bf16 = 50331648 B

  prep_kernel<<<64, 256, 0, stream>>>(w_qkv, w_out, pos, wsW, wsWo, wsBias);
  win_attn_kernel<<<2048, 128, 0, stream>>>(x, wsW, wsBias, wsAtt);
  out_proj_kernel<<<dim3(2048, 3), 128, 0, stream>>>(wsAtt, wsWo, b_out, out);
}
